// ScaledDotProductAttention_22874995818674
// MI455X (gfx1250) — matrix operand order repaired and checked by reading, not compiled, not run on hardware
//
#include <hip/hip_runtime.h>
#include <hip/hip_bf16.h>
#include <cstdint>

typedef __attribute__((ext_vector_type(16))) __bf16 v16bf;
typedef __attribute__((ext_vector_type(8)))  float  v8f;
typedef __attribute__((ext_vector_type(4)))  float  f4;

#define Bb 2
#define Hh 16
#define Ss 2048
#define Dk 64
#define SB_STRIDE 2052   // 2048 + 4 pad: breaks 8-row bank aliasing (8*4 % 64 != 0)
#define VST_STRIDE 68    // 64 + 4 pad for V staging

// A-fragment element -> K index map for 16-bit 16x32 A operand:
// element i of lane half h holds K = i + 16*h (matches B fill).
__device__ __forceinline__ int aoff(int i, int half) {
  return i + 16 * half;
}

__global__ void __launch_bounds__(256, 1)
sdpa_wmma_kernel(const float* __restrict__ Q, const float* __restrict__ K,
                 const float* __restrict__ V, const uint8_t* __restrict__ M,
                 float* __restrict__ ctx_out, float* __restrict__ attn_out) {
  extern __shared__ float smem[];
  float* Sb  = smem;                        // 16 x SB_STRIDE fp32 score stripe
  float* Vst = smem + 16 * SB_STRIDE;       // 8 waves x 32 x VST_STRIDE
  float* red = Vst + 8 * 32 * VST_STRIDE;   // 16 x 16 reduction scratch
  float* rst = red + 256;                   // [0..15] rowmax, [16..31] 1/rowsum

  const int tid  = threadIdx.x;
  const int w    = tid >> 5;       // wave 0..7
  const int lane = tid & 31;
  const int half = lane >> 4;      // lane half (WMMA operand layout)
  const int ln   = lane & 15;

  const int bh = blockIdx.x >> 7;  // (b*H + h), 0..31
  const int qt = blockIdx.x & 127;
  const int q0 = qt << 4;

  const float*   Qp = Q + (size_t)bh * Ss * Dk;
  const float*   Kp = K + (size_t)bh * Ss * Dk;
  const float*   Vp = V + (size_t)bh * Ss * Dk;
  const uint8_t* Mp = M + (size_t)bh * Ss * Ss;

  // ---------------- Phase 1: scores = (Q*0.125) @ K^T, mask, -> LDS ----------
  v16bf aq[2];  // Q fragments for both 32-wide d-chunks, scale folded in
  {
    const float* qr = Qp + (size_t)(q0 + ln) * Dk;
    #pragma unroll
    for (int c = 0; c < 2; ++c) {
      const int d0 = c * 32;
      v16bf a;
      #pragma unroll
      for (int i = 0; i < 16; ++i)
        a[i] = (__bf16)(0.125f * qr[d0 + aoff(i, half)]);
      aq[c] = a;
    }
  }

  for (int kt = 0; kt < 16; ++kt) {        // this wave's 16 key tiles
    const int k0 = (w << 8) + (kt << 4);
    v8f acc = {};
    #pragma unroll
    for (int c = 0; c < 2; ++c) {
      const int d0 = c * 32;
      // B operand (32x16): element i -> K = i + 16*half, N = ln (K^T column = key row)
      const float* kr = Kp + (size_t)(k0 + ln) * Dk + d0 + 16 * half;
      v16bf b;
      #pragma unroll
      for (int i = 0; i < 16; ++i) b[i] = (__bf16)kr[i];
      acc = __builtin_amdgcn_wmma_f32_16x16x32_bf16(false, aq[c], false, b,
                                                    (short)0, acc, false, false);
    }
    // D layout: VGPR vr -> row vr + 8*half, col = ln. Apply mask at store.
    // Mask is single-use streaming data: non-temporal load (TH=NT) so the
    // 134 MB mask doesn't evict the K/V reuse set from L2.
    #pragma unroll
    for (int vr = 0; vr < 8; ++vr) {
      const int r = vr + 8 * half;
      const int c = k0 + ln;
      const uint8_t m = __builtin_nontemporal_load(&Mp[(size_t)(q0 + r) * Ss + c]);
      Sb[r * SB_STRIDE + c] = m ? -1e9f : acc[vr];
    }
  }
  __syncthreads();

  // ---------------- Phase 2: row max (16 threads per row) --------------------
  const int rr  = tid >> 4;
  const int seg = tid & 15;
  {
    float m = -3.4e38f;
    #pragma unroll 4
    for (int j = 0; j < 128; ++j)
      m = fmaxf(m, Sb[rr * SB_STRIDE + seg + 16 * j]);
    red[rr * 16 + seg] = m;
  }
  __syncthreads();
  if (seg == 0) {
    float m = red[rr * 16];
    for (int j = 1; j < 16; ++j) m = fmaxf(m, red[rr * 16 + j]);
    rst[rr] = m;
  }
  __syncthreads();

  // ---------------- Phase 3: p = exp(s - max) (unnormalized), row sum --------
  {
    const float mx = rst[rr];
    float s = 0.f;
    #pragma unroll 4
    for (int j = 0; j < 128; ++j) {
      const int idx = rr * SB_STRIDE + seg + 16 * j;
      const float p = __expf(Sb[idx] - mx);
      Sb[idx] = p;
      s += p;
    }
    red[rr * 16 + seg] = s;
  }
  __syncthreads();
  if (seg == 0) {
    float s = 0.f;
    for (int j = 0; j < 16; ++j) s += red[rr * 16 + j];
    rst[16 + rr] = 1.0f / s;
  }
  __syncthreads();

  // ---------------- Phase 4: coalesced normalized attn write -----------------
  // attn output is write-once / never re-read: non-temporal store (TH=NT)
  // keeps the 537 MB stream out of L2, preserving K/V residency.
  {
    float* ap = attn_out + ((size_t)bh * Ss + q0) * Ss;
    #pragma unroll 4
    for (int j = 0; j < 128; ++j) {
      const int e = j * 256 + tid;          // row-major over 16x2048 tile
      const int r = e >> 11;
      const int c = e & 2047;
      __builtin_nontemporal_store(Sb[r * SB_STRIDE + c] * rst[16 + r],
                                  &ap[(size_t)r * Ss + c]);
    }
  }

  // ---------------- Phase 5: partial context = P_unnorm @ V ------------------
  v8f accv[4] = {};
  float* vst = Vst + w * 32 * VST_STRIDE;
  for (int ck = 0; ck < 8; ++ck) {          // this wave's 8 k-chunks of 32
    const int kb = (w << 8) + (ck << 5);
    // Stage V[kb..kb+31][0..63] into LDS, coalesced float4 loads (regular
    // temporal: V is re-read by 127 other q-tile blocks via L2).
    #pragma unroll
    for (int j = 0; j < 16; ++j) {
      const int e = lane + 32 * j;          // float4 index 0..511
      const int k = e >> 4, f = e & 15;
      const f4 v = *(const f4*)(Vp + (size_t)(kb + k) * Dk + f * 4);
      *(f4*)(vst + k * VST_STRIDE + f * 4) = v;
    }
    // A fragment from probabilities in LDS.
    v16bf a;
    #pragma unroll
    for (int i = 0; i < 16; ++i)
      a[i] = (__bf16)Sb[ln * SB_STRIDE + kb + aoff(i, half)];
    // 4 d-tiles of 16 columns each.
    #pragma unroll
    for (int dt = 0; dt < 4; ++dt) {
      v16bf b;
      #pragma unroll
      for (int i = 0; i < 16; ++i)
        b[i] = (__bf16)vst[(i + 16 * half) * VST_STRIDE + dt * 16 + ln];
      accv[dt] = __builtin_amdgcn_wmma_f32_16x16x32_bf16(false, a, false, b,
                                                         (short)0, accv[dt],
                                                         false, false);
    }
  }
  __syncthreads();                          // all waves done reading Sb

  // Dump per-wave partials into (reused) Sb region: [w][r*64 + d].
  {
    float* Cp = Sb + w * 1024;
    #pragma unroll
    for (int dt = 0; dt < 4; ++dt) {
      #pragma unroll
      for (int vr = 0; vr < 8; ++vr) {
        const int r = vr + 8 * half;
        const int d = dt * 16 + ln;
        Cp[r * 64 + d] = accv[dt][vr];
      }
    }
  }
  __syncthreads();

  // ---------------- Phase 6: reduce 8 partials, scale, write context ---------
  {
    float* cp = ctx_out + ((size_t)bh * Ss + q0) * Dk;
    #pragma unroll
    for (int j = 0; j < 4; ++j) {
      const int e = j * 256 + tid;          // 0..1023 = r*64 + d
      const int r = e >> 6;
      float s = 0.f;
      #pragma unroll
      for (int ww = 0; ww < 8; ++ww) s += Sb[ww * 1024 + e];
      __builtin_nontemporal_store(s * rst[16 + r], &cp[e]);
    }
  }
}

extern "C" void kernel_launch(void* const* d_in, const int* in_sizes, int n_in,
                              void* d_out, int out_size, void* d_ws, size_t ws_size,
                              hipStream_t stream) {
  (void)in_sizes; (void)n_in; (void)out_size; (void)d_ws; (void)ws_size;
  const float*   Q = (const float*)d_in[0];
  const float*   K = (const float*)d_in[1];
  const float*   V = (const float*)d_in[2];
  const uint8_t* M = (const uint8_t*)d_in[3];   // jnp.bool_ -> 1 byte/elem

  float* ctx  = (float*)d_out;                              // [B,H,S,DK]
  float* attn = ctx + (size_t)Bb * Hh * Ss * Dk;            // [B,H,S,S]

  const size_t shmem =
      (size_t)(16 * SB_STRIDE + 8 * 32 * VST_STRIDE + 256 + 32) * sizeof(float);

  static bool attr_set = false;  // idempotent, deterministic; not a stream op
  if (!attr_set) {
    hipFuncSetAttribute((const void*)sdpa_wmma_kernel,
                        hipFuncAttributeMaxDynamicSharedMemorySize, (int)shmem);
    attr_set = true;
  }

  dim3 grid(Bb * Hh * (Ss / 16));   // 4096 blocks: one 16-row q-tile each
  dim3 block(256);                  // 8 wave32s
  sdpa_wmma_kernel<<<grid, block, shmem, stream>>>(Q, K, V, M, ctx, attn);
}
